// CNN_LSTM_41205916238256
// MI455X (gfx1250) — compile-verified, matrix operand order checked by reading, not run on hardware
//
#include <hip/hip_runtime.h>

// ---------------------------------------------------------------------------
// CNN (conv-as-GEMM) + LSTM for MI455X / gfx1250, bf16 WMMA with f32 accum.
// gatesX staged in bf16 (halves dominant HBM stream); recurrent kernel
// double-buffers gate slabs into LDS with global_load_async_to_lds_b128.
// ---------------------------------------------------------------------------
static constexpr int VOCAB = 50257;
static constexpr int EMB   = 256;
static constexpr int KER   = 5;
static constexpr int NF    = 256;
static constexpr int HID   = 128;
static constexpr int BB    = 256;   // batch
static constexpr int SS    = 512;   // sequence
static constexpr int TT    = SS - KER + 1;   // 508
static constexpr int GATES = 4 * HID;        // 512
static constexpr int KC    = KER * EMB;      // 1280

typedef __attribute__((ext_vector_type(16))) __bf16 v16bf;
typedef __attribute__((ext_vector_type(8)))  float  v8f;

__device__ __forceinline__ v8f wmma_bf16(v16bf a, v16bf b, v8f c) {
    // (neg_a, A, neg_b, B, c_mod, C, reuse_a, reuse_b)
    return __builtin_amdgcn_wmma_f32_16x16x32_bf16(false, a, false, b,
                                                   (short)0, c, false, false);
}

// Async 16-byte global -> LDS copy (gfx1250), tracked by ASYNCcnt.
__device__ __forceinline__ void async_copy_b128(unsigned int lds_off,
                                                const void* gptr) {
    asm volatile("global_load_async_to_lds_b128 %0, %1, off"
                 :: "v"(lds_off), "v"((unsigned long long)(uintptr_t)gptr)
                 : "memory");
}
__device__ __forceinline__ void wait_async0() {
    asm volatile("s_wait_asynccnt 0" ::: "memory");
}

// A/B fragment from a row-major [N][ldk] bf16 matrix where lane (&15) selects
// row n (== column of B, or row of A), K per ISA 16-bit layout:
//   lanes 0-15 : K = kk+0..7,  kk+16..23
//   lanes 16-31: K = kk+8..15, kk+24..31
__device__ __forceinline__ v16bf frag_bf16(const __bf16* base, int ldk,
                                           int col, int lane, int kk) {
    const int n  = col + (lane & 15);
    const int hi = lane >> 4;
    const __bf16* p = base + (size_t)n * ldk + kk + hi * 8;
    union { v16bf v; uint4 u[2]; } r;
    r.u[0] = *(const uint4*)(p);        // K chunk 0 (16 bytes = 8 bf16)
    r.u[1] = *(const uint4*)(p + 16);   // K chunk 1
    return r.v;
}

// Same fragment but sourced from f32 memory, converted to bf16 on load.
__device__ __forceinline__ v16bf frag_f32(const float* base, int ldk,
                                          int col, int lane, int kk) {
    const int n  = col + (lane & 15);
    const int hi = lane >> 4;
    const float* p = base + (size_t)n * ldk + kk + hi * 8;
    union { v16bf v; __bf16 e[16]; } r;
#pragma unroll
    for (int i = 0; i < 8; ++i) {
        r.e[i]     = (__bf16)p[i];
        r.e[8 + i] = (__bf16)p[i + 16];
    }
    return r.v;
}

// A fragment for the conv GEMM: row m (= t within tile) of the im2col matrix,
// element K = k*EMB + e maps to embS[(m + k)][e].  8-element K-chunks never
// straddle an EMB boundary (kk % 8 == 0, EMB % 32 == 0).
__device__ __forceinline__ v16bf frag_embS(const __bf16* embS, int lane, int kk) {
    const int m  = lane & 15;
    const int hi = lane >> 4;
    const int K0 = kk + hi * 8;
    const int K1 = K0 + 16;
    union { v16bf v; uint4 u[2]; } r;
    r.u[0] = *(const uint4*)(embS + (m + (K0 >> 8)) * EMB + (K0 & 255));
    r.u[1] = *(const uint4*)(embS + (m + (K1 >> 8)) * EMB + (K1 & 255));
    return r.v;
}

__device__ __forceinline__ float sigmoidf(float x) {
    return 1.0f / (1.0f + __expf(-x));
}

// ---------------------------------------------------------------------------
// Kernel 1: convert conv weights ([NF][KER*EMB]) and w_ih ([512][256]) to bf16
// ---------------------------------------------------------------------------
__global__ void prep_weights(const float* __restrict__ conv_w,
                             const float* __restrict__ w_ih,
                             __bf16* __restrict__ W2,
                             __bf16* __restrict__ WIH) {
    const int n1 = NF * KC;         // 327680
    const int n2 = GATES * NF;      // 131072
    for (int idx = blockIdx.x * blockDim.x + threadIdx.x; idx < n1 + n2;
         idx += gridDim.x * blockDim.x) {
        if (idx < n1) W2[idx]       = (__bf16)conv_w[idx];
        else          WIH[idx - n1] = (__bf16)w_ih[idx - n1];
    }
}

// ---------------------------------------------------------------------------
// Kernel 2: fused  embed -> conv(+bias,ReLU) -> x @ w_ih^T (+b_ih+b_hh)
// grid = (TT/16 tiles, BB), block = 128 (4 waves).
// Stage 1: conv tile 16(t) x 256(f), K=1280 over LDS-staged embeddings.
// Stage 2: gates tile 16(t) x 512(g), K=256 with stage-1 output as LDS A.
// Gate pre-activations stored bf16 (biases folded) -> 133 MB stream.
// ---------------------------------------------------------------------------
__global__ void __launch_bounds__(128)
conv_gates(const int*   __restrict__ text,
           const float* __restrict__ emb,
           const float* __restrict__ conv_b,
           const float* __restrict__ b_ih,
           const float* __restrict__ b_hh,
           const __bf16* __restrict__ W2,    // [NF][1280]
           const __bf16* __restrict__ WIH,   // [512][256]
           __bf16* __restrict__ gatesX)      // [TT][BB][512] bf16
{
    __shared__ int    tokS[16 + KER - 1];          // 20 tokens
    __shared__ __bf16 embS[(16 + KER - 1) * EMB];  // 20 x 256 bf16 (10 KB)
    __shared__ __bf16 convS[16 * NF];              // 16 x 256 bf16 (8 KB)

    const int t0   = blockIdx.x * 16;
    const int b    = blockIdx.y;
    const int tid  = threadIdx.x;
    const int lane = tid & 31;
    const int wv   = tid >> 5;        // 0..3
    const int hi   = lane >> 4;

    if (tid < 20) {
        int row = t0 + tid;
        if (row > SS - 1) row = SS - 1;           // clamp tail tile
        tokS[tid] = text[(size_t)b * SS + row];
    }
    __syncthreads();
    for (int idx = tid; idx < 20 * EMB; idx += 128) {
        const int r = idx >> 8, e = idx & 255;
        embS[idx] = (__bf16)emb[(size_t)tokS[r] * EMB + e];
    }
    __syncthreads();

    // ---- stage 1: conv as 16x256 GEMM, K = 1280 ----
    v8f zero = {};
    v8f acc[4];
#pragma unroll
    for (int j = 0; j < 4; ++j) acc[j] = zero;

    for (int kk = 0; kk < KC; kk += 32) {
        const v16bf aF = frag_embS(embS, lane, kk);
#pragma unroll
        for (int j = 0; j < 4; ++j) {
            const v16bf bF = frag_bf16(W2, KC, (wv * 4 + j) * 16, lane, kk);
            acc[j] = wmma_bf16(aF, bF, acc[j]);
        }
    }

    // bias + ReLU -> convS (stage-2 A matrix, row-major [16][256])
#pragma unroll
    for (int j = 0; j < 4; ++j) {
        const int   f  = (wv * 4 + j) * 16 + (lane & 15);
        const float cb = conv_b[f];
#pragma unroll
        for (int r = 0; r < 8; ++r) {
            const int m = r + hi * 8;
            float v = acc[j][r] + cb;
            v = v > 0.0f ? v : 0.0f;
            convS[m * NF + f] = (__bf16)v;
        }
    }
    __syncthreads();

    // ---- stage 2: gates = relu(conv) @ w_ih^T, 16x512, K = 256 ----
    v8f acc2[8];
#pragma unroll
    for (int j = 0; j < 8; ++j) acc2[j] = zero;

    for (int kk = 0; kk < NF; kk += 32) {
        const v16bf aF = frag_bf16(convS, NF, 0, lane, kk);
#pragma unroll
        for (int j = 0; j < 8; ++j) {
            const v16bf bF = frag_bf16(WIH, NF, (wv * 8 + j) * 16, lane, kk);
            acc2[j] = wmma_bf16(aF, bF, acc2[j]);
        }
    }

#pragma unroll
    for (int j = 0; j < 8; ++j) {
        const int   g    = (wv * 8 + j) * 16 + (lane & 15);
        const float bias = b_ih[g] + b_hh[g];
#pragma unroll
        for (int r = 0; r < 8; ++r) {
            const int m = r + hi * 8;
            const int t = t0 + m;
            if (t < TT)
                gatesX[((size_t)t * BB + b) * GATES + g] =
                    (__bf16)(acc2[j][r] + bias);
        }
    }
}

// ---------------------------------------------------------------------------
// Kernel 3: LSTM recurrence.  Batch rows are independent: 16 workgroups of
// one 16-batch M-tile each, 8 waves (256 threads).  h lives in LDS (8 KB f32);
// each wave holds ALL its w_hh B-fragments in VGPRs (16 frags) for 508 steps.
// The 16 KB bf16 gate slab for step t+1 is double-buffered into LDS with
// global_load_async_to_lds_b128 (ASYNCcnt) while step t computes.
// ---------------------------------------------------------------------------
__global__ void __launch_bounds__(256)
lstm_rec(const __bf16* __restrict__ gatesX,  // [TT][BB][512] bf16, bias folded
         const float*  __restrict__ h0,      // [BB][HID]
         const float*  __restrict__ w_hh,    // [512][128]
         float*        __restrict__ out)     // [BB][HID]
{
    __shared__ float  hS[16 * HID];          // 8 KB
    __shared__ __bf16 gS[2][16 * GATES];     // 2 x 16 KB gate slabs

    const int tid  = threadIdx.x;
    const int lane = tid & 31;
    const int wv   = tid >> 5;               // 0..7 -> hid tile
    const int hi   = lane >> 4;
    const int b0   = blockIdx.x * 16;

    for (int idx = tid; idx < 16 * HID; idx += 256)
        hS[idx] = h0[(size_t)(b0 + (idx >> 7)) * HID + (idx & 127)];

    // Preload w_hh fragments: gate type x K-step, column block = gt*128+wv*16
    v16bf bFr[4][4];
#pragma unroll
    for (int gt = 0; gt < 4; ++gt)
#pragma unroll
        for (int kx = 0; kx < 4; ++kx)
            bFr[gt][kx] = frag_f32(w_hh, HID, gt * HID + wv * 16, lane, kx * 32);

    // Prologue: async-stage slab for t=0 (16 KB contiguous: rows b0..b0+15).
    {
        const __bf16* src = gatesX + (size_t)b0 * GATES;
#pragma unroll
        for (int kchunk = 0; kchunk < 4; ++kchunk) {
            const int idx = tid + kchunk * 256;          // 0..1023, x8 bf16
            async_copy_b128((unsigned int)(uintptr_t)&gS[0][idx * 8],
                            src + idx * 8);
        }
    }
    v8f cc = {};                             // cell state fragment (f32)
    float hn[8];
    wait_async0();
    __syncthreads();

    for (int t = 0; t < TT; ++t) {
        // kick off async staging of slab t+1 into the other buffer
        if (t + 1 < TT) {
            const __bf16* src = gatesX + ((size_t)(t + 1) * BB + b0) * GATES;
            __bf16* dst = &gS[(t + 1) & 1][0];
#pragma unroll
            for (int kchunk = 0; kchunk < 4; ++kchunk) {
                const int idx = tid + kchunk * 256;
                async_copy_b128((unsigned int)(uintptr_t)(dst + idx * 8),
                                src + idx * 8);
            }
        }

        // init accumulators from LDS-staged input-side gates
        const __bf16* slab = &gS[t & 1][0];
        v8f acc[4];
#pragma unroll
        for (int gt = 0; gt < 4; ++gt) {
            const int g = gt * HID + wv * 16 + (lane & 15);
#pragma unroll
            for (int r = 0; r < 8; ++r)
                acc[gt][r] = (float)slab[(r + hi * 8) * GATES + g];
        }

        // h @ w_hh^T : K = 128 in 4 steps, A from LDS with f32->bf16 on load
#pragma unroll
        for (int kx = 0; kx < 4; ++kx) {
            const v16bf aF = frag_f32(hS, HID, 0, lane, kx * 32);
#pragma unroll
            for (int gt = 0; gt < 4; ++gt)
                acc[gt] = wmma_bf16(aF, bFr[gt][kx], acc[gt]);
        }

        // elementwise LSTM cell
#pragma unroll
        for (int r = 0; r < 8; ++r) {
            const float iv = sigmoidf(acc[0][r]);
            const float fv = sigmoidf(acc[1][r]);
            const float gv = tanhf(acc[2][r]);
            const float ov = sigmoidf(acc[3][r]);
            const float cn = fv * cc[r] + iv * gv;
            cc[r] = cn;
            hn[r] = ov * tanhf(cn);
        }

        __syncthreads();                     // all waves done reading old h
        const int col = wv * 16 + (lane & 15);
#pragma unroll
        for (int r = 0; r < 8; ++r)
            hS[(r + hi * 8) * HID + col] = hn[r];
        wait_async0();                       // own slab copies landed
        __syncthreads();                     // everyone's copies + h visible
    }

    const int col = wv * 16 + (lane & 15);
#pragma unroll
    for (int r = 0; r < 8; ++r)
        out[(size_t)(b0 + r + hi * 8) * HID + col] = hn[r];
}

// ---------------------------------------------------------------------------
extern "C" void kernel_launch(void* const* d_in, const int* in_sizes, int n_in,
                              void* d_out, int out_size, void* d_ws, size_t ws_size,
                              hipStream_t stream) {
    const int*   text   = (const int*)  d_in[0];
    const float* h0     = (const float*)d_in[1];
    const float* emb    = (const float*)d_in[2];
    const float* conv_w = (const float*)d_in[3];
    const float* conv_b = (const float*)d_in[4];
    const float* w_ih   = (const float*)d_in[5];
    const float* w_hh   = (const float*)d_in[6];
    const float* b_ih   = (const float*)d_in[7];
    const float* b_hh   = (const float*)d_in[8];
    float*       out    = (float*)d_out;

    // workspace layout (256 B aligned regions)
    char*   ws         = (char*)d_ws;
    size_t  gatesBytes = (size_t)TT * BB * GATES * sizeof(__bf16);  // ~127 MiB
    size_t  off        = (gatesBytes + 255) & ~(size_t)255;
    __bf16* gatesX = (__bf16*)ws;
    __bf16* W2     = (__bf16*)(ws + off);
    off += ((size_t)NF * KC * sizeof(__bf16) + 255) & ~(size_t)255;
    __bf16* WIH    = (__bf16*)(ws + off);

    prep_weights<<<256, 256, 0, stream>>>(conv_w, w_ih, W2, WIH);

    dim3 gConv((TT + 15) / 16, BB);          // 32 x 256 workgroups
    conv_gates<<<gConv, 128, 0, stream>>>(text, emb, conv_b, b_ih, b_hh,
                                          W2, WIH, gatesX);

    lstm_rec<<<BB / 16, 256, 0, stream>>>(gatesX, h0, w_hh, out);
}
